// GNNMatrixMultiHeadAttention_66529043415202
// MI455X (gfx1250) — matrix operand order repaired and checked by reading, not compiled, not run on hardware
//
#include <hip/hip_runtime.h>
#include <hip/hip_bf16.h>

// ---------------------------------------------------------------------------
// GNN multi-head attention, fused flash-style, bf16 WMMA on gfx1250 (wave32).
// B=2, L=1024, D=256, H=8, dq=32, NL=4.
// ---------------------------------------------------------------------------

typedef __attribute__((ext_vector_type(16))) __bf16 v16bf;
typedef __attribute__((ext_vector_type(8)))  float  v8f;

#define BD   2
#define LL   1024
#define DD   256
#define NH   8
#define DQ   32
#define NLB  4

// Native f32 -> bf16 conversion (hardware v_cvt, packs in pairs) instead of
// the 3-VALU-op software RNE sequence.
__device__ __forceinline__ __bf16 f2bf(float f) { return (__bf16)f; }
__device__ __forceinline__ unsigned short bf16bits(float f) {
    __bf16 h = (__bf16)f;
    return __builtin_bit_cast(unsigned short, h);
}
// load two consecutive bf16 (stored as u16) into fragment slots idx, idx+1
__device__ __forceinline__ void ldpair(const unsigned short* p, v16bf& v, int idx) {
    unsigned int w = *(const unsigned int*)p;
    v[idx]     = __builtin_bit_cast(__bf16, (unsigned short)(w & 0xFFFFu));
    v[idx + 1] = __builtin_bit_cast(__bf16, (unsigned short)(w >> 16));
}
__device__ __forceinline__ float f4comp(const float4& v, int l) {
    switch (l) { case 0: return v.x; case 1: return v.y; case 2: return v.z; default: return v.w; }
}
// A-fragment K base for vgpr-pair vg (0..7), lane-half hi (0/1): contiguous K = 16*hi + 2*vg
__device__ __forceinline__ int aKbase(int vg, int hi) {
    return (hi << 4) + (vg << 1);   // {0,2,...,14} + 16*hi
}

// ---------------------------------------------------------------------------
// Kernel 1: projections.  blockDim=32 (one wave), grid=(128*16 tiles, 5).
//   lz==0 : query = q @ w_q^T + b_q          -> qry  (bf16, row-major [2048][256])
//   lz>=1 : ek_l  = k @ w_ks[l]^T            -> ek   (bf16, [l][2048][256])
//                                            -> ekT  (bf16, [l][b][h][32][1024])
// ---------------------------------------------------------------------------
__global__ void gnn_proj_kernel(const float* __restrict__ q,
                                const float* __restrict__ kk_in,
                                const float* __restrict__ w_q,
                                const float* __restrict__ b_q,
                                const float* __restrict__ w_ks,
                                unsigned short* __restrict__ qry,
                                unsigned short* __restrict__ ek,
                                unsigned short* __restrict__ ekT) {
    const int lz   = blockIdx.y;             // 0=query, 1..4 = label+1
    const int tile = blockIdx.x;             // 0..2047
    const int mt = tile >> 4, nt = tile & 15;
    const int m0 = mt << 4, n0 = nt << 4;
    const int lane = threadIdx.x & 31;
    const int n = lane & 15, hi = lane >> 4;

    const float* X    = (lz == 0) ? q : kk_in;
    const float* W    = (lz == 0) ? w_q : (w_ks + (size_t)(lz - 1) * DD * DD);
    const bool   hasB = (lz == 0);

    v8f c = {};
    #pragma unroll
    for (int kblk = 0; kblk < DD; kblk += 32) {
        v16bf a, bm;
        #pragma unroll
        for (int vg = 0; vg < 8; ++vg) {
            const int ka = aKbase(vg, hi);
            float2 av = *(const float2*)&X[(size_t)(m0 + (lane & 15)) * DD + kblk + ka];
            a[2 * vg]     = f2bf(av.x);
            a[2 * vg + 1] = f2bf(av.y);
            const int kb = (hi << 4) + (vg << 1);            // B: K = 16*hi + 2*vg
            float2 bv = *(const float2*)&W[(size_t)(n0 + n) * DD + kblk + kb];
            bm[2 * vg]     = f2bf(bv.x);
            bm[2 * vg + 1] = f2bf(bv.y);
        }
        c = __builtin_amdgcn_wmma_f32_16x16x32_bf16(false, a, false, bm, (short)0, c, false, false);
    }

    const float bias = hasB ? b_q[n0 + n] : 0.0f;
    #pragma unroll
    for (int r = 0; r < 8; ++r) {
        const int m = m0 + r + hi * 8;                       // global row (b*L + seq)
        const unsigned short h16 = bf16bits(c[r] + bias);
        if (lz == 0) {
            qry[(size_t)m * DD + n0 + n] = h16;
        } else {
            const int l = lz - 1;
            ek[((size_t)l * (BD * LL) + m) * DD + n0 + n] = h16;
            const int b  = m >> 10, seq = m & 1023;
            const int hh = n0 >> 5,  dd = (n0 & 16) + n;     // 16-wide tile stays in one head-half
            ekT[((((size_t)l * BD + b) * NH + hh) * DQ + dd) * LL + seq] = h16;
        }
    }
}

// ---------------------------------------------------------------------------
// Kernel 2: fused flash attention.  blockDim=32, grid=(64 qtiles, 8 heads, 2 b).
// One wave owns a 16-query x 32-dq output tile; loops over 32 k-tiles of 32.
// ---------------------------------------------------------------------------
__global__ void gnn_attn_kernel(const unsigned short* __restrict__ qry,
                                const unsigned short* __restrict__ ek,
                                const unsigned short* __restrict__ ekT,
                                const float* __restrict__ edge_mask,
                                const int*   __restrict__ pad_mask,
                                unsigned short* __restrict__ aout) {
    const int qt = blockIdx.x, h = blockIdx.y, b = blockIdx.z;
    const int q0 = qt << 4;
    const int lane = threadIdx.x & 31;
    const int n = lane & 15, hi = lane >> 4;

    __shared__ unsigned short pt[16 * 32];                   // P-tile transpose buffer

    // Q A-fragment (16 q-rows x 32 d), contiguous bf16 pairs
    v16bf qa;
    #pragma unroll
    for (int vg = 0; vg < 8; ++vg) {
        const int ka = aKbase(vg, hi);
        ldpair(&qry[(size_t)(b * LL + q0 + (lane & 15)) * DD + h * DQ + ka], qa, 2 * vg);
    }

    v8f acc0 = {}, acc1 = {};
    float mrun[8], srun[8];
    #pragma unroll
    for (int r = 0; r < 8; ++r) { mrun[r] = -1e30f; srun[r] = 0.0f; }

    const float4* em4 = (const float4*)edge_mask;            // [b][q][k] -> float4 over labels
    const float invt = 0.17677669529663687f;                 // 1/sqrt(32)

    for (int kt = 0; kt < LL / 32; ++kt) {
        const int k0 = kt << 5;

        // prefetch next k-tile's edge_mask rows (emits global_prefetch_b8)
        if (kt + 1 < LL / 32)
            __builtin_prefetch(&em4[((size_t)(b * LL + q0 + (lane & 15)) * LL) + k0 + 32], 0, 0);

        // --- edge masks for this tile, kept in registers ---
        float4 msk[2][8];
        #pragma unroll
        for (int f = 0; f < 2; ++f)
            #pragma unroll
            for (int r = 0; r < 8; ++r) {
                const int qrow = q0 + r + hi * 8, kcol = k0 + f * 16 + n;
                msk[f][r] = em4[(size_t)(b * LL + qrow) * LL + kcol];
            }

        // --- scores: aw = sum_l (Q @ Ek_l^T) .* mask_l ---
        v8f aw0 = {}, aw1 = {};
        #pragma unroll
        for (int l = 0; l < NLB; ++l) {
            const unsigned short* eb = ek + ((size_t)l * (BD * LL) + b * LL) * DD;
            v16bf bk0, bk1;
            #pragma unroll
            for (int vg = 0; vg < 8; ++vg) {
                const int d = (hi << 4) + (vg << 1);         // B: K=d
                ldpair(&eb[(size_t)(k0 + n) * DD + h * DQ + d],      bk0, 2 * vg);
                ldpair(&eb[(size_t)(k0 + 16 + n) * DD + h * DQ + d], bk1, 2 * vg);
            }
            v8f z = {};
            v8f s0 = __builtin_amdgcn_wmma_f32_16x16x32_bf16(false, qa, false, bk0, (short)0, z, false, false);
            v8f s1 = __builtin_amdgcn_wmma_f32_16x16x32_bf16(false, qa, false, bk1, (short)0, z, false, false);
            #pragma unroll
            for (int r = 0; r < 8; ++r) {
                aw0[r] += s0[r] * f4comp(msk[0][r], l);
                aw1[r] += s1[r] * f4comp(msk[1][r], l);
            }
        }

        // --- padding -> logits ---
        float lg0[8], lg1[8];
        #pragma unroll
        for (int r = 0; r < 8; ++r) {
            const int qrow = q0 + r + hi * 8;
            const int p0i = pad_mask[(size_t)(b * LL + qrow) * LL + k0 + n];
            const int p1i = pad_mask[(size_t)(b * LL + qrow) * LL + k0 + 16 + n];
            lg0[r] = p0i ? -1e30f : aw0[r] * invt;
            lg1[r] = p1i ? -1e30f : aw1[r] * invt;
        }

        // --- online softmax: row reductions across the 16-lane half-groups ---
        float tm[8];
        #pragma unroll
        for (int r = 0; r < 8; ++r) tm[r] = fmaxf(lg0[r], lg1[r]);
        #pragma unroll
        for (int off = 1; off < 16; off <<= 1)
            #pragma unroll
            for (int r = 0; r < 8; ++r) tm[r] = fmaxf(tm[r], __shfl_xor(tm[r], off, 32));

        float corr[8], p0[8], p1[8], rs[8];
        #pragma unroll
        for (int r = 0; r < 8; ++r) {
            const float mn = fmaxf(mrun[r], tm[r]);
            corr[r] = __expf(mrun[r] - mn);
            mrun[r] = mn;
            p0[r] = __expf(lg0[r] - mn);
            p1[r] = __expf(lg1[r] - mn);
            rs[r] = p0[r] + p1[r];
        }
        #pragma unroll
        for (int off = 1; off < 16; off <<= 1)
            #pragma unroll
            for (int r = 0; r < 8; ++r) rs[r] += __shfl_xor(rs[r], off, 32);
        #pragma unroll
        for (int r = 0; r < 8; ++r) {
            srun[r] = srun[r] * corr[r] + rs[r];
            acc0[r] *= corr[r];
            acc1[r] *= corr[r];
        }

        // --- aggregation: acc += (P .* mask_l) @ Ek_l  (LDS transpose C->A frag) ---
        #pragma unroll
        for (int l = 0; l < NLB; ++l) {
            #pragma unroll
            for (int f = 0; f < 2; ++f)
                #pragma unroll
                for (int r = 0; r < 8; ++r) {
                    const int m = r + hi * 8;
                    const float v = (f ? p1[r] : p0[r]) * f4comp(msk[f][r], l);
                    pt[m * 32 + f * 16 + n] = bf16bits(v);
                }
            asm volatile("s_wait_dscnt 0" ::: "memory");     // wave-local LDS RAW fence
            v16bf pa;
            #pragma unroll
            for (int vg = 0; vg < 8; ++vg) {
                const int ka = aKbase(vg, hi);
                ldpair(&pt[(lane & 15) * 32 + ka], pa, 2 * vg);
            }
            asm volatile("s_wait_dscnt 0" ::: "memory");     // reads done before next label's writes

            const unsigned short* et = ekT + (((size_t)l * BD + b) * NH + h) * (DQ * LL);
            v16bf be0, be1;
            #pragma unroll
            for (int vg = 0; vg < 8; ++vg) {
                const int kseq = k0 + (hi << 4) + (vg << 1); // B: K = seq-k
                ldpair(&et[(size_t)n * LL + kseq],        be0, 2 * vg);   // N = d 0..15
                ldpair(&et[(size_t)(16 + n) * LL + kseq], be1, 2 * vg);   // N = d 16..31
            }
            acc0 = __builtin_amdgcn_wmma_f32_16x16x32_bf16(false, pa, false, be0, (short)0, acc0, false, false);
            acc1 = __builtin_amdgcn_wmma_f32_16x16x32_bf16(false, pa, false, be1, (short)0, acc1, false, false);
        }
    }

    // --- finalize: divide by softmax denom, store bf16 head output ---
    #pragma unroll
    for (int r = 0; r < 8; ++r) {
        const float inv = 1.0f / srun[r];
        const int qrow = q0 + r + hi * 8;
        aout[(size_t)(b * LL + qrow) * DD + h * DQ + n]      = bf16bits(acc0[r] * inv);
        aout[(size_t)(b * LL + qrow) * DD + h * DQ + 16 + n] = bf16bits(acc1[r] * inv);
    }
}

// ---------------------------------------------------------------------------
// Kernel 3: output projection  out = aout @ w_h^T + b_h  (f32 result)
// ---------------------------------------------------------------------------
__global__ void gnn_outproj_kernel(const unsigned short* __restrict__ aout,
                                   const float* __restrict__ w_h,
                                   const float* __restrict__ b_h,
                                   float* __restrict__ out) {
    const int tile = blockIdx.x;
    const int mt = tile >> 4, nt = tile & 15;
    const int m0 = mt << 4, n0 = nt << 4;
    const int lane = threadIdx.x & 31;
    const int n = lane & 15, hi = lane >> 4;

    v8f c = {};
    #pragma unroll
    for (int kblk = 0; kblk < DD; kblk += 32) {
        v16bf a, bm;
        #pragma unroll
        for (int vg = 0; vg < 8; ++vg) {
            const int ka = aKbase(vg, hi);
            ldpair(&aout[(size_t)(m0 + (lane & 15)) * DD + kblk + ka], a, 2 * vg);
            const int kb = (hi << 4) + (vg << 1);
            float2 bv = *(const float2*)&w_h[(size_t)(n0 + n) * DD + kblk + kb];
            bm[2 * vg]     = f2bf(bv.x);
            bm[2 * vg + 1] = f2bf(bv.y);
        }
        c = __builtin_amdgcn_wmma_f32_16x16x32_bf16(false, a, false, bm, (short)0, c, false, false);
    }
    const float bias = b_h[n0 + n];
    #pragma unroll
    for (int r = 0; r < 8; ++r) {
        const int m = m0 + r + hi * 8;
        out[(size_t)m * DD + n0 + n] = c[r] + bias;
    }
}

// ---------------------------------------------------------------------------
// Host launcher
// ---------------------------------------------------------------------------
extern "C" void kernel_launch(void* const* d_in, const int* in_sizes, int n_in,
                              void* d_out, int out_size, void* d_ws, size_t ws_size,
                              hipStream_t stream) {
    const float* q         = (const float*)d_in[0];
    const float* k         = (const float*)d_in[1];
    const float* edge_mask = (const float*)d_in[2];
    const int*   pad_mask  = (const int*)  d_in[3];
    const float* w_q       = (const float*)d_in[4];
    const float* b_q       = (const float*)d_in[5];
    const float* w_ks      = (const float*)d_in[6];
    const float* w_h       = (const float*)d_in[7];
    const float* b_h       = (const float*)d_in[8];
    float* out = (float*)d_out;

    // workspace layout (bf16 as u16): query | ek | ekT | attn-out  (~10 MB)
    unsigned short* qry  = (unsigned short*)d_ws;
    unsigned short* ek   = qry + (size_t)BD * LL * DD;                 // 0.5M elems
    unsigned short* ekT  = ek  + (size_t)NLB * BD * LL * DD;           // 2M elems
    unsigned short* aout = ekT + (size_t)NLB * BD * NH * DQ * LL;      // 2M elems

    // 1) projections: 2048 tiles x (query + 4 labels)
    gnn_proj_kernel<<<dim3(2048, 5), 32, 0, stream>>>(q, k, w_q, b_q, w_ks, qry, ek, ekT);
    // 2) fused flash attention: (64 q-tiles, 8 heads, 2 batches)
    gnn_attn_kernel<<<dim3(64, 8, 2), 32, 0, stream>>>(qry, ek, ekT, edge_mask, pad_mask, aout);
    // 3) output projection
    gnn_outproj_kernel<<<dim3(2048), 32, 0, stream>>>(aout, w_h, b_h, out);
}
